// GraphReceiver_43447889166783
// MI455X (gfx1250) — matrix operand order repaired and checked by reading, not compiled, not run on hardware
//
#include <hip/hip_runtime.h>
#include <hip/hip_bf16.h>
#include <math.h>

#define N_NODES 10000
#define N_EDGES 160000
#define EDGE_DIM 9
#define KTOT 1056              // 32*32 weight rows + 32 bias rows
#define KSTRIDE 1064           // padded column stride (halves): 532 dwords = 20 mod 64 banks
#define NTILES (N_EDGES / 16)  // 10000 exactly

typedef __attribute__((ext_vector_type(16))) __bf16 v16bf;
typedef __attribute__((ext_vector_type(8)))  __bf16 v8bf;
typedef __attribute__((ext_vector_type(8)))  float  v8f;
typedef __attribute__((ext_vector_type(4)))  unsigned int v4u;

// ---------------------------------------------------------------------------
// Build T (bf16, column-major, padded stride KSTRIDE) from wb [32,1024]+bb[1024]
//   T[o*KSTRIDE + (k*32+i)] = wb[k][i*32+o],  T[o*KSTRIDE + 1024+i] = bb[i*32+o]
// ---------------------------------------------------------------------------
__global__ void build_T(const float* __restrict__ wb1, const float* __restrict__ bb1,
                        const float* __restrict__ wb2, const float* __restrict__ bb2,
                        __bf16* __restrict__ T1, __bf16* __restrict__ T2) {
  int idx = blockIdx.x * blockDim.x + threadIdx.x;
  const int per = 32 * KSTRIDE;
  if (idx >= 2 * per) return;
  int layer = idx / per;
  int rem   = idx - layer * per;
  int o = rem / KSTRIDE;
  int c = rem - o * KSTRIDE;
  const float* wb = layer ? wb2 : wb1;
  const float* bb = layer ? bb2 : bb1;
  float v = 0.0f;                       // pad region -> 0 (never read, but defined)
  if (c < 1024) {
    int k = c >> 5, i = c & 31;
    v = wb[k * 1024 + i * 32 + o];
  } else if (c < KTOT) {
    int i = c - 1024;
    v = bb[i * 32 + o];
  }
  __bf16* T = layer ? T2 : T1;
  T[o * KSTRIDE + c] = (__bf16)v;
}

__global__ void zero_f32(float* __restrict__ p, int n) {
  int i = blockIdx.x * blockDim.x + threadIdx.x;
  if (i < n) p[i] = 0.0f;
}

// ---------------------------------------------------------------------------
// Edge front: h_edge[e] = relu(edge_attr[e] @ wa + ba)  (bf16)
//             xs_edge[e] = nodes_in[src[e]]             (bf16)
//             cnt[dst[e]] += 1                          (degree for mean)
// ---------------------------------------------------------------------------
__global__ void edge_front(const float* __restrict__ edge_attr,
                           const int*  __restrict__ edge_index,   // [2, E]
                           const float* __restrict__ wa,          // [9, 32]
                           const float* __restrict__ ba,          // [32]
                           const float* __restrict__ nodes_in,    // [N, 32]
                           __bf16* __restrict__ h_edge,
                           __bf16* __restrict__ xs_edge,
                           float* __restrict__ cnt) {
  int t = blockIdx.x * blockDim.x + threadIdx.x;
  if (t >= N_EDGES * 32) return;
  int e = t >> 5, j = t & 31;
  float acc = ba[j];
#pragma unroll
  for (int d = 0; d < EDGE_DIM; ++d)
    acc += edge_attr[e * EDGE_DIM + d] * wa[d * 32 + j];
  acc = fmaxf(acc, 0.0f);
  h_edge[t] = (__bf16)acc;
  int src = edge_index[e];
  xs_edge[t] = (__bf16)nodes_in[src * 32 + j];
  if (j == 0) {
    int dst = edge_index[N_EDGES + e];
    atomicAdd(&cnt[dst], 1.0f);
  }
}

// ---------------------------------------------------------------------------
// Fused message GEMM + scatter. One wave per 16-edge tile; 8 waves/block.
//   msg[16,32] = A[16,1056] @ T[1056,32], A[e][k*32+i] = h[e][k]*x[e][i] (k<32),
//   A[e][1024+i] = x[e][i] (bias rows).  A built in registers each K-step.
// T is staged once per block into LDS (68 KB dynamic, gfx1250 has 320 KB/WGP);
// inner-loop B fragments come from ds_load_b128 (conflict-free via KSTRIDE pad).
// A-fragment lane layout (16-bit A, 16x32): lanes 0-15 -> K 0-15,
// lanes 16-31 -> K 16-31, row M = lane&15.
// C layout: lane = col (n), VGPR j = row M (lanes>=16: M=j+8) -> direct scatter.
// ---------------------------------------------------------------------------
__global__ __launch_bounds__(256)
void gemm_scatter(const __bf16* __restrict__ h_edge,
                  const __bf16* __restrict__ xs_edge,
                  const __bf16* __restrict__ T,
                  const int*   __restrict__ edge_index,
                  float* __restrict__ accum) {
  extern __shared__ __bf16 Tlds[];      // 32 * KSTRIDE halves = 68096 B

  // cooperative stage of T into LDS, 16B chunks (KSTRIDE*32 halves = 4256 v4u)
  {
    const v4u* src = (const v4u*)T;
    v4u* dst = (v4u*)Tlds;
    const int nchunks = (32 * KSTRIDE) / 8;   // 8 halves per 16B chunk
    for (int i = threadIdx.x; i < nchunks; i += 256) dst[i] = src[i];
  }
  __syncthreads();

  int lane = threadIdx.x & 31;
  int wv   = threadIdx.x >> 5;
  int tile = blockIdx.x * 8 + wv;
  if (tile >= NTILES) return;           // wave-uniform; grid covers exactly
  int half = (lane < 16) ? 0 : 1;
  int row  = tile * 16 + (lane & 15);

  int xoff = row * 32 + half * 16;
  v8bf xlo = *(const v8bf*)(xs_edge + xoff);        // K i-block 0..7 / 16..23
  v8bf xhi = *(const v8bf*)(xs_edge + xoff + 8);    // K i-block 8..15 / 24..31
  const __bf16* hrow = h_edge + row * 32;

  int n0 = lane & 15;
  const __bf16* Tb0 = Tlds + n0 * KSTRIDE + half * 16;
  const __bf16* Tb1 = Tlds + (n0 + 16) * KSTRIDE + half * 16;

  v8f c0 = {};
  v8f c1 = {};
  for (int kk = 0; kk < 33; ++kk) {
    float hs = (kk < 32) ? (float)hrow[kk] : 1.0f;  // kk==32: bias rows (h==1)
    v16bf a;
#pragma unroll
    for (int i = 0; i < 8; ++i) a[i]     = (__bf16)(hs * (float)xlo[i]);
#pragma unroll
    for (int i = 0; i < 8; ++i) a[8 + i] = (__bf16)(hs * (float)xhi[i]);
    v16bf b0 = *(const v16bf*)(Tb0 + kk * 32);
    v16bf b1 = *(const v16bf*)(Tb1 + kk * 32);
    c0 = __builtin_amdgcn_wmma_f32_16x16x32_bf16(false, a, false, b0,
                                                 (short)0, c0, false, false);
    c1 = __builtin_amdgcn_wmma_f32_16x16x32_bf16(false, a, false, b1,
                                                 (short)0, c1, false, false);
  }

  int mbase = tile * 16 + half * 8;
#pragma unroll
  for (int j = 0; j < 8; ++j) {
    int d = edge_index[N_EDGES + mbase + j];
    atomicAdd(&accum[d * 32 + n0],      c0[j]);
    atomicAdd(&accum[d * 32 + 16 + n0], c1[j]);
  }
}

// ---------------------------------------------------------------------------
// Node update: h_out = relu(accum/max(cnt,1) + nodes_in @ root + bias)
// ---------------------------------------------------------------------------
__global__ void node_update(const float* __restrict__ accum,
                            const float* __restrict__ cnt,
                            const float* __restrict__ nodes_in,
                            const float* __restrict__ root,   // [32,32]
                            const float* __restrict__ bias,   // [32]
                            float* __restrict__ h_out) {
  int t = blockIdx.x * blockDim.x + threadIdx.x;
  if (t >= N_NODES * 32) return;
  int v = t >> 5, j = t & 31;
  float r = bias[j];
#pragma unroll
  for (int i = 0; i < 32; ++i) r += nodes_in[v * 32 + i] * root[i * 32 + j];
  float c = fmaxf(cnt[v], 1.0f);
  h_out[t] = fmaxf(accum[t] / c + r, 0.0f);
}

// ---------------------------------------------------------------------------
// scores = h2 @ emb[signal]; out = log_softmax(scores) over all N nodes.
// Single 1024-thread block (32 waves), shared-memory tree reductions.
// ---------------------------------------------------------------------------
__global__ __launch_bounds__(1024)
void final_softmax(const float* __restrict__ h2,
                   const float* __restrict__ emb_table,
                   const int*  __restrict__ signal,
                   float* __restrict__ scores,
                   float* __restrict__ out) {
  __shared__ float sig[32];
  __shared__ float red[32];
  __shared__ float sM, sS;
  int tid = threadIdx.x;
  if (tid < 32) sig[tid] = emb_table[signal[0] * 32 + tid];
  __syncthreads();

  float lmax = -3.0e38f;
  for (int v = tid; v < N_NODES; v += 1024) {
    float s = 0.0f;
#pragma unroll
    for (int i = 0; i < 32; ++i) s += h2[v * 32 + i] * sig[i];
    scores[v] = s;
    lmax = fmaxf(lmax, s);
  }
  int lane = tid & 31, wid = tid >> 5;
  for (int off = 16; off > 0; off >>= 1) lmax = fmaxf(lmax, __shfl_xor(lmax, off, 32));
  if (lane == 0) red[wid] = lmax;
  __syncthreads();
  if (wid == 0) {
    float m = red[lane];
    for (int off = 16; off > 0; off >>= 1) m = fmaxf(m, __shfl_xor(m, off, 32));
    if (lane == 0) sM = m;
  }
  __syncthreads();
  float M = sM;

  float lsum = 0.0f;
  for (int v = tid; v < N_NODES; v += 1024) lsum += expf(scores[v] - M);
  for (int off = 16; off > 0; off >>= 1) lsum += __shfl_xor(lsum, off, 32);
  __syncthreads();
  if (lane == 0) red[wid] = lsum;
  __syncthreads();
  if (wid == 0) {
    float s = red[lane];
    for (int off = 16; off > 0; off >>= 1) s += __shfl_xor(s, off, 32);
    if (lane == 0) sS = s;
  }
  __syncthreads();
  float lse = M + logf(sS);
  for (int v = tid; v < N_NODES; v += 1024) out[v] = scores[v] - lse;
}

// ---------------------------------------------------------------------------
extern "C" void kernel_launch(void* const* d_in, const int* in_sizes, int n_in,
                              void* d_out, int out_size, void* d_ws, size_t ws_size,
                              hipStream_t stream) {
  const int*   signal     = (const int*)  d_in[0];
  const float* x          = (const float*)d_in[1];
  const int*   edge_index = (const int*)  d_in[2];
  const float* edge_attr  = (const float*)d_in[3];
  const float* w1a = (const float*)d_in[4];
  const float* b1a = (const float*)d_in[5];
  const float* w1b = (const float*)d_in[6];
  const float* b1b = (const float*)d_in[7];
  const float* root1 = (const float*)d_in[8];
  const float* bias1 = (const float*)d_in[9];
  const float* w2a = (const float*)d_in[10];
  const float* b2a = (const float*)d_in[11];
  const float* w2b = (const float*)d_in[12];
  const float* b2b = (const float*)d_in[13];
  const float* root2 = (const float*)d_in[14];
  const float* bias2 = (const float*)d_in[15];
  const float* emb_table = (const float*)d_in[16];
  float* out = (float*)d_out;

  // workspace carve-out (256B aligned), ~25 MB total
  unsigned char* w = (unsigned char*)d_ws;
  auto carve = [&](size_t bytes) {
    unsigned char* p = w;
    w += (bytes + 255) & ~(size_t)255;
    return p;
  };
  __bf16* T1      = (__bf16*)carve((size_t)32 * KSTRIDE * sizeof(__bf16));
  __bf16* T2      = (__bf16*)carve((size_t)32 * KSTRIDE * sizeof(__bf16));
  __bf16* h_edge  = (__bf16*)carve((size_t)N_EDGES * 32 * sizeof(__bf16));
  __bf16* xs_edge = (__bf16*)carve((size_t)N_EDGES * 32 * sizeof(__bf16));
  float*  accum   = (float*) carve((size_t)N_NODES * 32 * sizeof(float));
  float*  cnt     = (float*) carve((size_t)N_NODES * sizeof(float));
  float*  h1      = (float*) carve((size_t)N_NODES * 32 * sizeof(float));
  float*  h2      = (float*) carve((size_t)N_NODES * 32 * sizeof(float));
  float*  scores  = (float*) carve((size_t)N_NODES * sizeof(float));

  const size_t ldsT = (size_t)32 * KSTRIDE * sizeof(__bf16);  // 68096 B

  build_T<<<(2 * 32 * KSTRIDE + 255) / 256, 256, 0, stream>>>(w1b, b1b, w2b, b2b, T1, T2);

  // ---- layer 1 (input x -> h1) ----
  zero_f32<<<(N_NODES * 32 + 255) / 256, 256, 0, stream>>>(accum, N_NODES * 32);
  zero_f32<<<(N_NODES + 255) / 256, 256, 0, stream>>>(cnt, N_NODES);
  edge_front<<<(N_EDGES * 32 + 255) / 256, 256, 0, stream>>>(
      edge_attr, edge_index, w1a, b1a, x, h_edge, xs_edge, cnt);
  gemm_scatter<<<NTILES / 8, 256, ldsT, stream>>>(h_edge, xs_edge, T1, edge_index, accum);
  node_update<<<(N_NODES * 32 + 255) / 256, 256, 0, stream>>>(
      accum, cnt, x, root1, bias1, h1);

  // ---- layer 2 (input h1 -> h2) ----
  zero_f32<<<(N_NODES * 32 + 255) / 256, 256, 0, stream>>>(accum, N_NODES * 32);
  zero_f32<<<(N_NODES + 255) / 256, 256, 0, stream>>>(cnt, N_NODES);
  edge_front<<<(N_EDGES * 32 + 255) / 256, 256, 0, stream>>>(
      edge_attr, edge_index, w2a, b2a, h1, h_edge, xs_edge, cnt);
  gemm_scatter<<<NTILES / 8, 256, ldsT, stream>>>(h_edge, xs_edge, T2, edge_index, accum);
  node_update<<<(N_NODES * 32 + 255) / 256, 256, 0, stream>>>(
      accum, cnt, h1, root2, bias2, h2);

  // ---- readout ----
  final_softmax<<<1, 1024, 0, stream>>>(h2, emb_table, signal, scores, out);
}
